// LSTMPricePredictor_67783173866126
// MI455X (gfx1250) — matrix operand order repaired and checked by reading, not compiled, not run on hardware
//
#include <hip/hip_runtime.h>

typedef __attribute__((ext_vector_type(16))) _Float16   v16h;
typedef __attribute__((ext_vector_type(8)))  float      v8f;
typedef __attribute__((ext_vector_type(4)))  unsigned   v4u;

namespace {
constexpr int kH   = 50;    // hidden size
constexpr int kT   = 512;   // sequence length
constexpr int kB   = 4096;  // batch
constexpr int kRows = 32;   // batch rows per workgroup
constexpr int kThreads = 128; // 4 wave32 waves
constexpr int kHS  = 136;   // hA row stride in halves ([h0|h1] = 128 + 8 pad)
constexpr int kTC  = 32;    // timesteps staged per LDS chunk

constexpr int kWB0Halves = 2 * 16 * 512;   // layer0 Whh fragments (K=64,  N=256)
constexpr int kWB1Halves = 4 * 16 * 512;   // layer1 [Wih|Whh]     (K=128, N=256)
constexpr int kHAHalves  = kRows * kHS;
constexpr int kXCFloats  = kRows * kTC;
constexpr size_t kSmemBytes =
    size_t(kWB0Halves + kWB1Halves + kHAHalves) * 2 +
    size_t(kXCFloats + 3 * 256) * 4;                  // = 114176 B
}

union FragCast { v4u q[2]; v16h h; };

// B fragment (32x16 f16, K x N): lane L holds K = (h/8)*16 + (L/16)*8 + h%8, N = L%16.
// Stored pre-packed lane-major: frag*512 + lane*16 halves -> two ds_load_b128.
__device__ inline v16h load_frag_B(const _Float16* wb, int fragIdx, int lane) {
  const _Float16* p = wb + fragIdx * 512 + lane * 16;
  FragCast f;
  f.q[0] = *(const v4u*)(p);
  f.q[1] = *(const v4u*)(p + 8);
  return f.h;
}

// A fragment (16x32 f16, M x K): lane L holds M = L%16,
// K = kbase + (L/16)*8 + {0..7} and +16+{0..7}  -> two ds_load_b128 per lane.
__device__ inline v16h load_frag_A(const _Float16* hA, int row, int kbase) {
  const _Float16* p = hA + row * kHS + kbase;
  FragCast f;
  f.q[0] = *(const v4u*)(p);
  f.q[1] = *(const v4u*)(p + 16);
  return f.h;
}

__device__ inline float sigm_(float v) { return 1.0f / (1.0f + __expf(-v)); }
__device__ inline float tanh_(float v) {
  v = fminf(fmaxf(v, -15.0f), 15.0f);
  const float e = __expf(2.0f * v);
  return (e - 1.0f) / (e + 1.0f);
}

__global__ void __launch_bounds__(kThreads)
lstm2_fused(const float* __restrict__ x,
            const float* __restrict__ Wih0, const float* __restrict__ Whh0,
            const float* __restrict__ bih0, const float* __restrict__ bhh0,
            const float* __restrict__ Wih1, const float* __restrict__ Whh1,
            const float* __restrict__ bih1, const float* __restrict__ bhh1,
            const float* __restrict__ W1, const float* __restrict__ b1,
            const float* __restrict__ W2, const float* __restrict__ b2,
            const float* __restrict__ W3, const float* __restrict__ b3,
            float* __restrict__ out)
{
  extern __shared__ unsigned char smem[];
  _Float16* wB0 = (_Float16*)smem;
  _Float16* wB1 = wB0 + kWB0Halves;
  _Float16* hA  = wB1 + kWB1Halves;            // [kRows][kHS]: cols 0-63 h0, 64-127 h1
  float*    xc    = (float*)(hA + kHAHalves);  // [kRows][kTC]
  float*    wih0c = xc + kXCFloats;            // [256]
  float*    bias0 = wih0c + 256;               // [256]
  float*    bias1 = bias0 + 256;               // [256]

  const int tid   = threadIdx.x;
  const int bbase = blockIdx.x * kRows;

  // ---- pack weights into WMMA B-fragment layout in LDS (once) ----
  for (int idx = tid; idx < kWB0Halves; idx += kThreads) {
    const int frag = idx >> 9, rem = idx & 511;
    const int lane = rem >> 4, hh = rem & 15;
    const int kc = frag >> 4, nt = frag & 15;
    const int K = kc * 32 + (hh >> 3) * 16 + (lane >> 4) * 8 + (hh & 7);
    const int N = nt * 16 + (lane & 15);
    const int g = N >> 6, j = N & 63;
    float v = 0.0f;
    if (j < kH && K < kH) v = Whh0[(g * kH + j) * kH + K];  // B[k][n] = Whh[n_row, k]
    wB0[idx] = (_Float16)v;
  }
  for (int idx = tid; idx < kWB1Halves; idx += kThreads) {
    const int frag = idx >> 9, rem = idx & 511;
    const int lane = rem >> 4, hh = rem & 15;
    const int kc = frag >> 4, nt = frag & 15;
    const int K = kc * 32 + (hh >> 3) * 16 + (lane >> 4) * 8 + (hh & 7);          // 0..127: [h0|h1]
    const int N = nt * 16 + (lane & 15);
    const int g = N >> 6, j = N & 63;
    float v = 0.0f;
    if (j < kH) {
      if (K < 64) { if (K < kH)       v = Wih1[(g * kH + j) * kH + K]; }
      else        { const int k2 = K - 64; if (k2 < kH) v = Whh1[(g * kH + j) * kH + k2]; }
    }
    wB1[idx] = (_Float16)v;
  }
  for (int n = tid; n < 256; n += kThreads) {
    const int g = n >> 6, j = n & 63;
    float w = 0.0f, bb0 = 0.0f, bb1 = 0.0f;
    if (j < kH) {
      w   = Wih0[g * kH + j];
      bb0 = bih0[g * kH + j] + bhh0[g * kH + j];
      bb1 = bih1[g * kH + j] + bhh1[g * kH + j];
    }
    wih0c[n] = w; bias0[n] = bb0; bias1[n] = bb1;
  }
  for (int i = tid; i < kHAHalves; i += kThreads) hA[i] = (_Float16)0.0f;
  __syncthreads();

  const int wv   = tid >> 5;
  const int lane = tid & 31;
  const int m    = wv >> 1;        // batch tile (rows m*16 .. m*16+15)
  const int jt   = wv & 1;         // j-column half (j in [jt*32, jt*32+32))
  const int lhi  = lane >> 4, llo = lane & 15;
  const int arow = m * 16 + llo;   // A-matrix row for this lane (M = lane%16)

  // cell state in registers, per C-layout element: b = m*16 + lhi*8 + r,
  // j = jt*32 + jj*16 + llo
  float c0[2][8], c1[2][8];
  #pragma unroll
  for (int jj = 0; jj < 2; ++jj)
    #pragma unroll
    for (int r = 0; r < 8; ++r) { c0[jj][r] = 0.0f; c1[jj][r] = 0.0f; }

  float bc0[4][2], bc1[4][2], wx0[4][2];
  #pragma unroll
  for (int g = 0; g < 4; ++g)
    #pragma unroll
    for (int jj = 0; jj < 2; ++jj) {
      const int col = g * 64 + jt * 32 + jj * 16 + llo;
      bc0[g][jj] = bias0[col];
      bc1[g][jj] = bias1[col];
      wx0[g][jj] = wih0c[col];
    }
  __syncthreads();

  for (int tc = 0; tc < kT / kTC; ++tc) {
    if (tc + 1 < kT / kTC) {        // prefetch next x chunk (global_prefetch_b8)
      for (int k = tid; k < kRows; k += kThreads)
        __builtin_prefetch(&x[size_t(bbase + k) * kT + (tc + 1) * kTC], 0, 0);
    }
    __syncthreads();
    for (int k = tid; k < kRows * kTC; k += kThreads) {
      const int b = k >> 5, tt = k & 31;
      xc[b * kTC + tt] = x[size_t(bbase + b) * kT + tc * kTC + tt];
    }
    __syncthreads();

    for (int tt = 0; tt < kTC; ++tt) {
      // ================= layer 0: gates = h0 @ Whh0^T =================
      v8f acc[4][2];
      #pragma unroll
      for (int g = 0; g < 4; ++g)
        #pragma unroll
        for (int jj = 0; jj < 2; ++jj) acc[g][jj] = v8f{};
      #pragma unroll
      for (int kc = 0; kc < 2; ++kc) {
        const v16h a = load_frag_A(hA, arow, kc * 32 + lhi * 8);
        #pragma unroll
        for (int g = 0; g < 4; ++g)
          #pragma unroll
          for (int jj = 0; jj < 2; ++jj) {
            const int nt = g * 4 + jt * 2 + jj;
            const v16h bm = load_frag_B(wB0, kc * 16 + nt, lane);
            acc[g][jj] = __builtin_amdgcn_wmma_f32_16x16x32_f16(
                false, a, false, bm, (short)0, acc[g][jj], false, false);
          }
      }
      // -------- layer0 LSTM cell (in-register, fp32) --------
      _Float16 h0n[2][8];
      #pragma unroll
      for (int jj = 0; jj < 2; ++jj)
        #pragma unroll
        for (int r = 0; r < 8; ++r) {
          const int bl = m * 16 + lhi * 8 + r;
          const float xv = xc[bl * kTC + tt];
          const float gi = acc[0][jj][r] + bc0[0][jj] + xv * wx0[0][jj];
          const float gf = acc[1][jj][r] + bc0[1][jj] + xv * wx0[1][jj];
          const float gg = acc[2][jj][r] + bc0[2][jj] + xv * wx0[2][jj];
          const float go = acc[3][jj][r] + bc0[3][jj] + xv * wx0[3][jj];
          const float c = sigm_(gf) * c0[jj][r] + sigm_(gi) * tanh_(gg);
          c0[jj][r] = c;
          h0n[jj][r] = (_Float16)(sigm_(go) * tanh_(c));
        }
      __syncthreads();   // all layer-0 A reads done -> safe to overwrite h0
      #pragma unroll
      for (int jj = 0; jj < 2; ++jj)
        #pragma unroll
        for (int r = 0; r < 8; ++r)
          hA[(m * 16 + lhi * 8 + r) * kHS + jt * 32 + jj * 16 + llo] = h0n[jj][r];
      __syncthreads();   // h0(t) visible

      // ============ layer 1: gates = [h0(t)|h1(t-1)] @ [Wih1|Whh1]^T ============
      #pragma unroll
      for (int g = 0; g < 4; ++g)
        #pragma unroll
        for (int jj = 0; jj < 2; ++jj) acc[g][jj] = v8f{};
      #pragma unroll
      for (int kc = 0; kc < 4; ++kc) {
        const v16h a = load_frag_A(hA, arow, kc * 32 + lhi * 8);
        #pragma unroll
        for (int g = 0; g < 4; ++g)
          #pragma unroll
          for (int jj = 0; jj < 2; ++jj) {
            const int nt = g * 4 + jt * 2 + jj;
            const v16h bm = load_frag_B(wB1, kc * 16 + nt, lane);
            acc[g][jj] = __builtin_amdgcn_wmma_f32_16x16x32_f16(
                false, a, false, bm, (short)0, acc[g][jj], false, false);
          }
      }
      _Float16 h1n[2][8];
      #pragma unroll
      for (int jj = 0; jj < 2; ++jj)
        #pragma unroll
        for (int r = 0; r < 8; ++r) {
          const float gi = acc[0][jj][r] + bc1[0][jj];
          const float gf = acc[1][jj][r] + bc1[1][jj];
          const float gg = acc[2][jj][r] + bc1[2][jj];
          const float go = acc[3][jj][r] + bc1[3][jj];
          const float c = sigm_(gf) * c1[jj][r] + sigm_(gi) * tanh_(gg);
          c1[jj][r] = c;
          h1n[jj][r] = (_Float16)(sigm_(go) * tanh_(c));
        }
      __syncthreads();   // all layer-1 A reads done -> safe to overwrite h1
      #pragma unroll
      for (int jj = 0; jj < 2; ++jj)
        #pragma unroll
        for (int r = 0; r < 8; ++r)
          hA[(m * 16 + lhi * 8 + r) * kHS + 64 + jt * 32 + jj * 16 + llo] = h1n[jj][r];
      __syncthreads();   // h1(t) visible
    }
  }

  // ================= MLP head on final h1 (tiny, fp32 VALU) =================
  if (tid < kRows) {
    const int b = tid;
    float hv[kH];
    for (int k = 0; k < kH; ++k) hv[k] = (float)hA[b * kHS + 64 + k];
    float a1[32];
    #pragma unroll 4
    for (int o = 0; o < 32; ++o) {
      float s = b1[o];
      for (int k = 0; k < kH; ++k) s += hv[k] * W1[o * kH + k];
      a1[o] = s > 0.0f ? s : 0.0f;
    }
    float a2[16];
    #pragma unroll 4
    for (int o = 0; o < 16; ++o) {
      float s = b2[o];
      for (int k = 0; k < 32; ++k) s += a1[k] * W2[o * 32 + k];
      a2[o] = s > 0.0f ? s : 0.0f;
    }
    float s = b3[0];
    for (int k = 0; k < 16; ++k) s += a2[k] * W3[k];
    out[bbase + b] = s;
  }
}

extern "C" void kernel_launch(void* const* d_in, const int* in_sizes, int n_in,
                              void* d_out, int out_size, void* d_ws, size_t ws_size,
                              hipStream_t stream) {
  (void)in_sizes; (void)n_in; (void)d_ws; (void)ws_size; (void)out_size;
  const float* x    = (const float*)d_in[0];
  const float* Wih0 = (const float*)d_in[1];
  const float* Whh0 = (const float*)d_in[2];
  const float* bih0 = (const float*)d_in[3];
  const float* bhh0 = (const float*)d_in[4];
  const float* Wih1 = (const float*)d_in[5];
  const float* Whh1 = (const float*)d_in[6];
  const float* bih1 = (const float*)d_in[7];
  const float* bhh1 = (const float*)d_in[8];
  const float* W1   = (const float*)d_in[9];
  const float* b1   = (const float*)d_in[10];
  const float* W2   = (const float*)d_in[11];
  const float* b2   = (const float*)d_in[12];
  const float* W3   = (const float*)d_in[13];
  const float* b3   = (const float*)d_in[14];

  dim3 grid(kB / kRows);   // 128 workgroups x 32 batch rows
  dim3 block(kThreads);    // 4 wave32 waves
  hipLaunchKernelGGL(lstm2_fused, grid, block, kSmemBytes, stream,
                     x, Wih0, Whh0, bih0, bhh0, Wih1, Whh1, bih1, bhh1,
                     W1, b1, W2, b2, W3, b3, (float*)d_out);
}
